// CycleGNN_9509057593730
// MI455X (gfx1250) — matrix operand order repaired and checked by reading, not compiled, not run on hardware
//
#include <hip/hip_runtime.h>
#include <math.h>

// ---------------- constants (match reference) ----------------
#define DD 64
#define LL 3
#define NRELS 474
#define EE 256000
#define NN 40000
#define NB 32
#define NT 64          // 2*B target edges
#define EPSF 1e-5f
#define LOG2E 1.4426950408889634f

// ---------------- types ----------------
typedef _Float16 v16h __attribute__((ext_vector_type(16)));
typedef _Float16 v8h  __attribute__((ext_vector_type(8)));
typedef float    v8f  __attribute__((ext_vector_type(8)));

// ---------------- WMMA helpers ----------------
__device__ inline v8f wmma16(v16h a, v16h b, v8f c) {
    return __builtin_amdgcn_wmma_f32_16x16x32_f16(false, a, false, b, (short)0, c, false, false);
}

// A-fragment (16x32 f16, M x K): lanes 0-15 rows, VGPR0-3 = K[hs*8 .. +7], VGPR4-7 = K[16+hs*8 .. +7]
__device__ inline v16h frag_a(const _Float16* row, int kbase, int hs) {
    union { v16h v; v8h p[2]; } u;
    u.p[0] = *(const v8h*)(row + kbase + hs * 8);
    u.p[1] = *(const v8h*)(row + kbase + 16 + hs * 8);
    return u.v;
}
// B-fragment (32x16 f16, K x N), stored transposed (N-major, K-minor): lane holds col n,
// VGPR0-3 = K[hs*8 .. +7], VGPR4-7 = K[16+hs*8 .. +7].
__device__ inline v16h frag_b(const _Float16* row, int kbase, int hs) {
    union { v16h v; v8h p[2]; } u;
    const _Float16* p = row + kbase + hs * 8;
    u.p[0] = *(const v8h*)(p);
    u.p[1] = *(const v8h*)(p + 16);
    return u.v;
}

// ---------------- fast nonlinearities: native v_exp_f32 (exp2) + v_rcp_f32 ----------------
__device__ inline float sigf(float x) {
    // 1/(1+e^-x) = rcp(1 + exp2(-x*log2e))
    return __builtin_amdgcn_rcpf(1.0f + __builtin_amdgcn_exp2f(-x * LOG2E));
}
__device__ inline float tanh_f(float x) {
    // tanh(x) = 2*sigmoid(2x) - 1
    return 2.0f * __builtin_amdgcn_rcpf(1.0f + __builtin_amdgcn_exp2f(-2.0f * x * LOG2E)) - 1.0f;
}

__device__ inline void atomic_max_f(float* a, float v) {
    if (v >= 0.0f) atomicMax((int*)a, __float_as_int(v));
    else           atomicMin((unsigned int*)a, __float_as_uint(v));
}
__device__ inline void atomic_min_f(float* a, float v) {
    if (v >= 0.0f) atomicMin((int*)a, __float_as_int(v));
    else           atomicMax((unsigned int*)a, __float_as_uint(v));
}

__device__ inline void ln_accum64(const float* xp, const float* g, const float* b, float* op) {
    float m = 0.f;
    for (int j = 0; j < DD; j++) m += xp[j];
    m *= (1.0f / DD);
    float v = 0.f;
    for (int j = 0; j < DD; j++) { float d = xp[j] - m; v += d * d; }
    v *= (1.0f / DD);
    float inv = rsqrtf(v + EPSF);
    for (int j = 0; j < DD; j++) op[j] += (xp[j] - m) * inv * g[j] + b[j];
}

// ---------------- utility kernels ----------------
__global__ void k_fill(float* __restrict__ p, long n, float v) {
    long i = (long)blockIdx.x * blockDim.x + threadIdx.x;
    if (i < n) p[i] = v;
}

__global__ void k_reset_aggs(float* s, float* mx, float* mn, float* sq, long n) {
    long i = (long)blockIdx.x * blockDim.x + threadIdx.x;
    if (i < n) { s[i] = 0.f; sq[i] = 0.f; mx[i] = -INFINITY; mn[i] = INFINITY; }
}

// dst[n*K + k] = src[k*Nn + n]  (f32 -> f16, transpose)
__global__ void k_transpose_f16(const float* __restrict__ src, _Float16* __restrict__ dst, int K, int Nn) {
    int i = blockIdx.x * blockDim.x + threadIdx.x;
    if (i >= K * Nn) return;
    int k = i / Nn, n = i - k * Nn;
    dst[(size_t)n * K + k] = (_Float16)src[i];
}

__global__ void k_equery_base(const int* __restrict__ etype, const int* __restrict__ tei,
                              const float* __restrict__ qemb, const float* __restrict__ ew,
                              const float* __restrict__ eb, float* __restrict__ base) {
    int idx = blockIdx.x * blockDim.x + threadIdx.x;
    if (idx >= NB * DD) return;
    int g = idx >> 6, j = idx & 63;
    float acc = eb[j];
    for (int k = 0; k < 2 * DD; k++) {
        int slot = 2 * g + (k >= DD);
        int et = etype[tei[slot]];
        acc += qemb[(size_t)et * DD + (k & 63)] * ew[k * DD + j];
    }
    base[idx] = acc;
}

__global__ void k_init_edges(const int* __restrict__ gid, const float* __restrict__ base,
                             float* __restrict__ efeat, float* __restrict__ equery, long total) {
    long i = (long)blockIdx.x * blockDim.x + threadIdx.x;
    if (i >= total) return;
    int e = (int)(i >> 6), j = (int)(i & 63);
    efeat[i]  = 0.f;
    equery[i] = base[gid[e] * DD + j];
}

__global__ void k_scatter_targets(const int* __restrict__ tei, const int* __restrict__ etype,
                                  const float* __restrict__ qemb, float* __restrict__ efeat) {
    int t = blockIdx.x, j = threadIdx.x;
    int e = tei[t];
    efeat[(size_t)e * DD + j] = qemb[(size_t)etype[e] * DD + j];
}

__global__ void k_degrees(const int* __restrict__ src, const int* __restrict__ dst,
                          float* __restrict__ od, float* __restrict__ id, int E) {
    int i = blockIdx.x * blockDim.x + threadIdx.x;
    if (i >= E) return;
    atomicAdd(&od[src[i]], 1.0f);
    atomicAdd(&id[dst[i]], 1.0f);
}

__global__ void k_avg_sum(const float* __restrict__ od, float* __restrict__ dsum, int n) {
    __shared__ float red[256];
    int i = blockIdx.x * 256 + threadIdx.x;
    red[threadIdx.x] = (i < n) ? logf(od[i] + 1.0f) : 0.0f;
    __syncthreads();
    for (int s = 128; s > 0; s >>= 1) {
        if (threadIdx.x < s) red[threadIdx.x] += red[threadIdx.x + s];
        __syncthreads();
    }
    if (threadIdx.x == 0) atomicAdd(dsum, red[0]);
}

__global__ void k_scales(const float* __restrict__ dsum, const float* __restrict__ ind,
                         float* __restrict__ dsafe, float* __restrict__ amp, float* __restrict__ att, int n) {
    int i = blockIdx.x * blockDim.x + threadIdx.x;
    if (i >= n) return;
    float avg = dsum[0] / (float)n;
    float di = ind[i];
    dsafe[i] = fmaxf(di, 1.0f);
    float ld = logf(di + 1.0f);
    amp[i] = ld / avg;
    att[i] = (ld > 0.0f) ? (avg / fmaxf(ld, EPSF)) : 0.0f;
}

// ---------------- layer kernel 1: GRU edge messages + segment atomics ----------------
// 128 edges/block, 8 waves x 16 edges each.  gx = x@Wx, gh = h@Wh via WMMA (K=64 -> 2 chunks),
// N-tiles grouped so r (jt), z (jt+4), n (jt+8) align per-lane.
__global__ __launch_bounds__(256) void k_gru_msg(
    const int* __restrict__ src, const int* __restrict__ dst, const int* __restrict__ etype,
    const float* __restrict__ nfeat, const float* __restrict__ equery, const float* __restrict__ efeat,
    const float* __restrict__ relw,
    const _Float16* __restrict__ WxT, const _Float16* __restrict__ WhT,
    const float* __restrict__ bx, const float* __restrict__ bh,
    float* __restrict__ aggSum, float* __restrict__ aggMax,
    float* __restrict__ aggMin, float* __restrict__ aggSq) {
    __shared__ _Float16 sX[128 * DD];
    __shared__ _Float16 sH[128 * DD];
    __shared__ float    sM[128 * DD];
    const int t = threadIdx.x;
    const int e0 = blockIdx.x * 128;
    for (int i = t; i < 128 * DD; i += 256) {
        int r = i >> 6, j = i & 63;
        int e = e0 + r;
        float x = nfeat[(size_t)src[e] * DD + j] + equery[(size_t)e * DD + j];
        float h = efeat[(size_t)e * DD + j] * relw[(size_t)etype[e] * DD + j];
        sX[i] = (_Float16)x;
        sH[i] = (_Float16)h;
    }
    __syncthreads();
    const int lane = t & 31, wave = t >> 5;
    const int lrow = lane & 15, hs = lane >> 4, rb = wave * 16;
    const _Float16* xr = sX + (rb + lrow) * DD;
    const _Float16* hr = sH + (rb + lrow) * DD;
    v16h ax0 = frag_a(xr, 0, hs), ax1 = frag_a(xr, 32, hs);
    v16h ah0 = frag_a(hr, 0, hs), ah1 = frag_a(hr, 32, hs);
#pragma unroll
    for (int jt = 0; jt < 4; jt++) {
        v8f aR = {}, aZ = {}, aXN = {}, aHN = {};
        {
            const _Float16* w  = WxT + (size_t)(jt * 16 + lrow) * DD;
            const _Float16* wh = WhT + (size_t)(jt * 16 + lrow) * DD;
            aR = wmma16(ax0, frag_b(w, 0, hs), aR);  aR = wmma16(ax1, frag_b(w, 32, hs), aR);
            aR = wmma16(ah0, frag_b(wh, 0, hs), aR); aR = wmma16(ah1, frag_b(wh, 32, hs), aR);
        }
        {
            const _Float16* w  = WxT + (size_t)((jt + 4) * 16 + lrow) * DD;
            const _Float16* wh = WhT + (size_t)((jt + 4) * 16 + lrow) * DD;
            aZ = wmma16(ax0, frag_b(w, 0, hs), aZ);  aZ = wmma16(ax1, frag_b(w, 32, hs), aZ);
            aZ = wmma16(ah0, frag_b(wh, 0, hs), aZ); aZ = wmma16(ah1, frag_b(wh, 32, hs), aZ);
        }
        {
            const _Float16* w  = WxT + (size_t)((jt + 8) * 16 + lrow) * DD;
            const _Float16* wh = WhT + (size_t)((jt + 8) * 16 + lrow) * DD;
            aXN = wmma16(ax0, frag_b(w, 0, hs), aXN);  aXN = wmma16(ax1, frag_b(w, 32, hs), aXN);
            aHN = wmma16(ah0, frag_b(wh, 0, hs), aHN); aHN = wmma16(ah1, frag_b(wh, 32, hs), aHN);
        }
        int col = jt * 16 + lrow;
        float brz = bx[col] + bh[col];
        float bzz = bx[DD + col] + bh[DD + col];
        float bxn = bx[2 * DD + col], bhn = bh[2 * DD + col];
#pragma unroll
        for (int v = 0; v < 8; v++) {
            int row = rb + hs * 8 + v;
            float r_ = sigf(aR[v] + brz);
            float z_ = sigf(aZ[v] + bzz);
            float n_ = tanh_f(aXN[v] + bxn + r_ * (aHN[v] + bhn));
            float hv = (float)sH[row * DD + col];
            sM[row * DD + col] = (1.0f - z_) * n_ + z_ * hv;
        }
    }
    __syncthreads();
    for (int i = t; i < 128 * DD; i += 256) {
        int r = i >> 6, j = i & 63;
        int d = dst[e0 + r];
        float v = sM[i];
        size_t o = (size_t)d * DD + j;
        atomicAdd(aggSum + o, v);
        atomicAdd(aggSq + o, v * v);
        atomic_max_f(aggMax + o, v);
        atomic_min_f(aggMin + o, v);
    }
}

// ---------------- layer kernel 2: PNA node update (feats[768] @ W -> 64) ----------------
// 64 nodes/block, 4 waves x 16 nodes.  K = 768 (24 chunks), 4 N-tiles.
__global__ __launch_bounds__(128) void k_pna(
    const float* __restrict__ aggSum, const float* __restrict__ aggMax,
    const float* __restrict__ aggMin, const float* __restrict__ aggSq,
    const float* __restrict__ in_deg, const float* __restrict__ dsafe,
    const float* __restrict__ s_amp, const float* __restrict__ s_att,
    const _Float16* __restrict__ WT, const float* __restrict__ pb,
    float* __restrict__ nfeat_new, int n_nodes) {
    __shared__ _Float16 sF[64 * 768];
    const int t = threadIdx.x;
    const int n0 = blockIdx.x * 64;
    for (int i = t; i < 64 * DD; i += 128) {
        int r = i >> 6, j = i & 63;
        int n = n0 + r;
        if (n >= n_nodes) n = n_nodes - 1;
        size_t o = (size_t)n * DD + j;
        float ds = dsafe[n];
        bool hm = in_deg[n] > 0.0f;
        float mean = aggSum[o] / ds;
        float mx = hm ? aggMax[o] : 0.0f;
        float mn = hm ? aggMin[o] : 0.0f;
        float var = aggSq[o] / ds - mean * mean;
        if (var < 0.0f) var = 0.0f;
        float sd = sqrtf(var + EPSF);
        float amp = s_amp[n], att = s_att[n];
        _Float16* f = sF + (size_t)r * 768;
        f[j]        = (_Float16)mean;        f[64 + j]  = (_Float16)mx;
        f[128 + j]  = (_Float16)mn;          f[192 + j] = (_Float16)sd;
        f[256 + j]  = (_Float16)(mean * amp); f[320 + j] = (_Float16)(mx * amp);
        f[384 + j]  = (_Float16)(mn * amp);   f[448 + j] = (_Float16)(sd * amp);
        f[512 + j]  = (_Float16)(mean * att); f[576 + j] = (_Float16)(mx * att);
        f[640 + j]  = (_Float16)(mn * att);   f[704 + j] = (_Float16)(sd * att);
    }
    __syncthreads();
    const int lane = t & 31, wave = t >> 5;
    const int lrow = lane & 15, hs = lane >> 4, rb = wave * 16;
    const _Float16* arow = sF + (size_t)(rb + lrow) * 768;
    v8f acc[4] = {};
    for (int kc = 0; kc < 24; kc++) {
        v16h a = frag_a(arow, kc * 32, hs);
#pragma unroll
        for (int nt = 0; nt < 4; nt++) {
            const _Float16* b = WT + (size_t)(nt * 16 + lrow) * 768;
            acc[nt] = wmma16(a, frag_b(b, kc * 32, hs), acc[nt]);
        }
    }
#pragma unroll
    for (int nt = 0; nt < 4; nt++) {
        int col = nt * 16 + lrow;
        float bb = pb[col];
#pragma unroll
        for (int v = 0; v < 8; v++) {
            int node = n0 + rb + hs * 8 + v;
            if (node < n_nodes) nfeat_new[(size_t)node * DD + col] = acc[nt][v] + bb;
        }
    }
}

__global__ void k_node_ln(const float* __restrict__ x, const float* __restrict__ g,
                          const float* __restrict__ b, float* __restrict__ accum, int R) {
    int r = blockIdx.x * blockDim.x + threadIdx.x;
    if (r >= R) return;
    ln_accum64(x + (size_t)r * DD, g, b, accum + (size_t)r * DD);
}

// ---------------- layer kernel 3: LSTM edge gates + fused LayerNorm accumulation ----------
__global__ __launch_bounds__(256) void k_lstm(
    const int* __restrict__ dst, const float* __restrict__ nfeat_new,
    const _Float16* __restrict__ WxT, const _Float16* __restrict__ WhT,
    const float* __restrict__ lb, const float* __restrict__ lng, const float* __restrict__ lnb,
    float* __restrict__ efeat, float* __restrict__ equery) {
    __shared__ _Float16 sA[128 * DD];
    __shared__ _Float16 sE[128 * DD];
    __shared__ float    sHH[128 * DD];
    __shared__ float    sC[128 * DD];
    const int t = threadIdx.x;
    const int e0 = blockIdx.x * 128;
    for (int i = t; i < 128 * DD; i += 256) {
        int r = i >> 6, j = i & 63;
        int e = e0 + r;
        sA[i] = (_Float16)nfeat_new[(size_t)dst[e] * DD + j];
        sE[i] = (_Float16)efeat[(size_t)e * DD + j];
    }
    __syncthreads();
    const int lane = t & 31, wave = t >> 5;
    const int lrow = lane & 15, hs = lane >> 4, rb = wave * 16;
    const _Float16* ar = sA + (rb + lrow) * DD;
    const _Float16* er = sE + (rb + lrow) * DD;
    v16h aa0 = frag_a(ar, 0, hs), aa1 = frag_a(ar, 32, hs);
    v16h ae0 = frag_a(er, 0, hs), ae1 = frag_a(er, 32, hs);
#pragma unroll
    for (int jt = 0; jt < 4; jt++) {
        v8f acc[4] = {};
#pragma unroll
        for (int gsel = 0; gsel < 4; gsel++) {
            int tile = jt + gsel * 4;
            const _Float16* wx = WxT + (size_t)(tile * 16 + lrow) * DD;
            acc[gsel] = wmma16(aa0, frag_b(wx, 0, hs), acc[gsel]);
            acc[gsel] = wmma16(aa1, frag_b(wx, 32, hs), acc[gsel]);
            const _Float16* wh = WhT + (size_t)(tile * 16 + lrow) * DD;
            acc[gsel] = wmma16(ae0, frag_b(wh, 0, hs), acc[gsel]);
            acc[gsel] = wmma16(ae1, frag_b(wh, 32, hs), acc[gsel]);
        }
        int col = jt * 16 + lrow;
        float bi = lb[col], bf = lb[DD + col], bg = lb[2 * DD + col], bo = lb[3 * DD + col];
#pragma unroll
        for (int v = 0; v < 8; v++) {
            int row = rb + hs * 8 + v;
            int e = e0 + row;
            float iv = sigf(acc[0][v] + bi);
            float fv = sigf(acc[1][v] + bf);
            float gv = tanh_f(acc[2][v] + bg);
            float ov = sigf(acc[3][v] + bo);
            float q = equery[(size_t)e * DD + col];
            float c = fv * q + iv * gv;
            sC[row * DD + col] = c;
            sHH[row * DD + col] = ov * tanh_f(c);
        }
    }
    __syncthreads();
    if (t < 128) {
        int e = e0 + t;
        ln_accum64(sHH + t * DD, lng, lnb, efeat + (size_t)e * DD);
        ln_accum64(sC + t * DD, lng, lnb, equery + (size_t)e * DD);
    }
}

__global__ void k_snapshot(const int* __restrict__ tei, const int* __restrict__ src,
                           const float* __restrict__ efeat, const float* __restrict__ equery,
                           const float* __restrict__ nfeat,
                           float* __restrict__ snapE, float* __restrict__ snapQ,
                           float* __restrict__ snapN, int l) {
    int t = blockIdx.x, j = threadIdx.x;
    int e = tei[t];
    size_t so = ((size_t)l * NT + t) * DD + j;
    snapE[so] = efeat[(size_t)e * DD + j];
    snapQ[so] = equery[(size_t)e * DD + j];
    snapN[so] = nfeat[(size_t)src[e] * DD + j];
}

// ---------------- final: JK projections (64 rows each) + scoring ----------------
__global__ __launch_bounds__(256) void k_final(
    const float* __restrict__ snapE, const float* __restrict__ snapQ, const float* __restrict__ snapN,
    const float* __restrict__ ejw, const float* __restrict__ ejb,
    const float* __restrict__ njw, const float* __restrict__ njb,
    const float* __restrict__ qjw, const float* __restrict__ qjb,
    const float* __restrict__ fcw, const float* __restrict__ fcb, float* __restrict__ out) {
    __shared__ float sE[NT * DD], sQ[NT * DD], sN[NT * DD];
    int t = threadIdx.x;
    for (int idx = t; idx < NT * DD; idx += 256) {
        int tt = idx >> 6, j = idx & 63;
        float ae = ejb[j], aq = qjb[j], an = njb[j];
        for (int k = 0; k < LL * DD; k++) {
            int l = k >> 6, jj = k & 63;
            size_t so = ((size_t)l * NT + tt) * DD + jj;
            ae += snapE[so] * ejw[k * DD + j];
            aq += snapQ[so] * qjw[k * DD + j];
            an += snapN[so] * njw[k * DD + j];
        }
        sE[idx] = ae; sQ[idx] = aq; sN[idx] = an;
    }
    __syncthreads();
    if (t < NB) {
        const float* e0p = sE + (2 * t) * DD;     const float* e1p = sE + (2 * t + 1) * DD;
        const float* q0p = sQ + (2 * t) * DD;     const float* q1p = sQ + (2 * t + 1) * DD;
        const float* h0  = sN + (2 * t) * DD;     const float* h1  = sN + (2 * t + 1) * DD;
        float r = fcb[0], lf = fcb[0];
        for (int k = 0; k < DD; k++) {
            r  += e0p[k] * fcw[k] + q0p[k] * fcw[DD + k] + h0[k] * fcw[2 * DD + k] + h1[k] * fcw[3 * DD + k];
            lf += e1p[k] * fcw[k] + q1p[k] * fcw[DD + k] + h1[k] * fcw[2 * DD + k] + h0[k] * fcw[3 * DD + k];
        }
        out[t] = fmaxf(r, lf);
    }
}

// ---------------- host orchestration ----------------
extern "C" void kernel_launch(void* const* d_in, const int* in_sizes, int n_in,
                              void* d_out, int out_size, void* d_ws, size_t ws_size,
                              hipStream_t stream) {
    const int* src   = (const int*)d_in[0];
    const int* dst   = (const int*)d_in[1];
    const int* etype = (const int*)d_in[2];
    const int* gid   = (const int*)d_in[3];
    const int* tei   = (const int*)d_in[4];
    const float* qemb   = (const float*)d_in[6];
    const float* eqp_w  = (const float*)d_in[7];
    const float* eqp_b  = (const float*)d_in[8];
    const float* rel_w  = (const float*)d_in[9];
    const float* gru_wx = (const float*)d_in[10];
    const float* gru_wh = (const float*)d_in[11];
    const float* gru_bx = (const float*)d_in[12];
    const float* gru_bh = (const float*)d_in[13];
    const float* pna_w  = (const float*)d_in[14];
    const float* pna_b  = (const float*)d_in[15];
    const float* lstm_wx = (const float*)d_in[16];
    const float* lstm_wh = (const float*)d_in[17];
    const float* lstm_b  = (const float*)d_in[18];
    const float* ln_g = (const float*)d_in[19];
    const float* ln_b = (const float*)d_in[20];
    const float* ejk_w = (const float*)d_in[21];
    const float* ejk_b = (const float*)d_in[22];
    const float* njk_w = (const float*)d_in[23];
    const float* njk_b = (const float*)d_in[24];
    const float* qjk_w = (const float*)d_in[25];
    const float* qjk_b = (const float*)d_in[26];
    const float* fc_w = (const float*)d_in[27];
    const float* fc_b = (const float*)d_in[28];

    const int E = in_sizes[0];   // 256000 (multiple of 128)
    const int N = NN;            // reference constant n_nodes = 40000

    // ---- workspace carve-up (256B aligned) ----
    size_t off = 0;
    char* wsb = (char*)d_ws;
    auto take = [&](size_t bytes) -> char* {
        char* p = wsb + off;
        off = (off + bytes + 255) & ~(size_t)255;
        return p;
    };
    float* efeat     = (float*)take((size_t)E * DD * 4);
    float* equery    = (float*)take((size_t)E * DD * 4);
    float* nfeat     = (float*)take((size_t)N * DD * 4);
    float* nfeat_new = (float*)take((size_t)N * DD * 4);
    float* aggSum    = (float*)take((size_t)N * DD * 4);
    float* aggMax    = (float*)take((size_t)N * DD * 4);
    float* aggMin    = (float*)take((size_t)N * DD * 4);
    float* aggSq     = (float*)take((size_t)N * DD * 4);
    float* out_deg   = (float*)take((size_t)N * 4);
    float* in_deg    = (float*)take((size_t)N * 4);
    float* dsafe     = (float*)take((size_t)N * 4);
    float* s_amp     = (float*)take((size_t)N * 4);
    float* s_att     = (float*)take((size_t)N * 4);
    float* dsum      = (float*)take(256);
    float* qbase     = (float*)take((size_t)NB * DD * 4);
    float* snapE     = (float*)take((size_t)LL * NT * DD * 4);
    float* snapQ     = (float*)take((size_t)LL * NT * DD * 4);
    float* snapN     = (float*)take((size_t)LL * NT * DD * 4);
    _Float16* gWxT = (_Float16*)take((size_t)LL * 3 * DD * DD * 2);  // [L][192][64]
    _Float16* gWhT = (_Float16*)take((size_t)LL * 3 * DD * DD * 2);
    _Float16* lWxT = (_Float16*)take((size_t)LL * 4 * DD * DD * 2);  // [L][256][64]
    _Float16* lWhT = (_Float16*)take((size_t)LL * 4 * DD * DD * 2);
    _Float16* pWT  = (_Float16*)take((size_t)LL * DD * 12 * DD * 2); // [L][64][768]

    // ---- weight conversion/transposition to f16 ----
    for (int l = 0; l < LL; l++) {
        k_transpose_f16<<<(3 * DD * DD + 255) / 256, 256, 0, stream>>>(
            gru_wx + (size_t)l * DD * 3 * DD, gWxT + (size_t)l * 3 * DD * DD, DD, 3 * DD);
        k_transpose_f16<<<(3 * DD * DD + 255) / 256, 256, 0, stream>>>(
            gru_wh + (size_t)l * DD * 3 * DD, gWhT + (size_t)l * 3 * DD * DD, DD, 3 * DD);
        k_transpose_f16<<<(4 * DD * DD + 255) / 256, 256, 0, stream>>>(
            lstm_wx + (size_t)l * DD * 4 * DD, lWxT + (size_t)l * 4 * DD * DD, DD, 4 * DD);
        k_transpose_f16<<<(4 * DD * DD + 255) / 256, 256, 0, stream>>>(
            lstm_wh + (size_t)l * DD * 4 * DD, lWhT + (size_t)l * 4 * DD * DD, DD, 4 * DD);
        k_transpose_f16<<<(12 * DD * DD + 255) / 256, 256, 0, stream>>>(
            pna_w + (size_t)l * 12 * DD * DD, pWT + (size_t)l * DD * 12 * DD, 12 * DD, DD);
    }

    // ---- setup ----
    k_fill<<<((long)N * DD + 255) / 256, 256, 0, stream>>>(nfeat, (long)N * DD, 0.f);
    k_fill<<<(N + 255) / 256, 256, 0, stream>>>(out_deg, N, 0.f);
    k_fill<<<(N + 255) / 256, 256, 0, stream>>>(in_deg, N, 0.f);
    k_fill<<<1, 256, 0, stream>>>(dsum, 1, 0.f);
    k_equery_base<<<(NB * DD + 255) / 256, 256, 0, stream>>>(etype, tei, qemb, eqp_w, eqp_b, qbase);
    k_init_edges<<<((long)E * DD + 255) / 256, 256, 0, stream>>>(gid, qbase, efeat, equery, (long)E * DD);
    k_scatter_targets<<<NT, DD, 0, stream>>>(tei, etype, qemb, efeat);
    k_degrees<<<(E + 255) / 256, 256, 0, stream>>>(src, dst, out_deg, in_deg, E);
    k_avg_sum<<<(N + 255) / 256, 256, 0, stream>>>(out_deg, dsum, N);
    k_scales<<<(N + 255) / 256, 256, 0, stream>>>(dsum, in_deg, dsafe, s_amp, s_att, N);

    // ---- layers ----
    for (int l = 0; l < LL; l++) {
        k_reset_aggs<<<((long)N * DD + 255) / 256, 256, 0, stream>>>(aggSum, aggMax, aggMin, aggSq, (long)N * DD);
        k_gru_msg<<<E / 128, 256, 0, stream>>>(
            src, dst, etype, nfeat, equery, efeat,
            rel_w + (size_t)l * NRELS * DD,
            gWxT + (size_t)l * 3 * DD * DD, gWhT + (size_t)l * 3 * DD * DD,
            gru_bx + (size_t)l * 3 * DD, gru_bh + (size_t)l * 3 * DD,
            aggSum, aggMax, aggMin, aggSq);
        k_pna<<<(N + 63) / 64, 128, 0, stream>>>(
            aggSum, aggMax, aggMin, aggSq, in_deg, dsafe, s_amp, s_att,
            pWT + (size_t)l * DD * 12 * DD, pna_b + (size_t)l * DD, nfeat_new, N);
        k_node_ln<<<(N + 255) / 256, 256, 0, stream>>>(
            nfeat_new, ln_g + (size_t)l * DD, ln_b + (size_t)l * DD, nfeat, N);
        k_lstm<<<E / 128, 256, 0, stream>>>(
            dst, nfeat_new,
            lWxT + (size_t)l * 4 * DD * DD, lWhT + (size_t)l * 4 * DD * DD,
            lstm_b + (size_t)l * 4 * DD, ln_g + (size_t)l * DD, ln_b + (size_t)l * DD,
            efeat, equery);
        k_snapshot<<<NT, DD, 0, stream>>>(tei, src, efeat, equery, nfeat, snapE, snapQ, snapN, l);
    }

    // ---- JK + scoring ----
    k_final<<<1, 256, 0, stream>>>(snapE, snapQ, snapN, ejk_w, ejk_b, njk_w, njk_b,
                                   qjk_w, qjk_b, fc_w, fc_b, (float*)d_out);
}
